// ExcitedMACE_36825049596261
// MI455X (gfx1250) — compile-verified, matrix operand order checked by reading, not run on hardware
//
#include <hip/hip_runtime.h>
#include <hip/hip_bf16.h>

typedef __attribute__((ext_vector_type(16))) _Float16 v16h;
typedef __attribute__((ext_vector_type(8)))  _Float16 v8h;
typedef __attribute__((ext_vector_type(8)))  float    v8f;

#define PI_F 3.14159265358979f
#define RMAXF 5.0f

__device__ __forceinline__ float siluf(float x){ float s = 1.f/(1.f+__expf(-x)); return x*s; }
__device__ __forceinline__ float dsiluf(float x){ float s = 1.f/(1.f+__expf(-x)); return s*(1.f + x*(1.f-s)); }

__device__ const int c_lmap[16] = {0,1,1,1,2,2,2,2,2,3,3,3,3,3,3,3};

__device__ __forceinline__ int speciesOf(const float* attrs, int n){
  int sp = 0;
#pragma unroll
  for (int e = 0; e < 10; ++e) if (attrs[n*10+e] > 0.5f) sp = e;
  return sp;
}

// ---------------------------------------------------------------------------
// Wave-level 16x16 GEMM tile: C[16,16] += A[16,K]*B[K,16], f16 in / f32 acc.
// A: row-major in LDS; per-lane fragment = two contiguous 16B spans.
// B: pre-swizzled into fragment order -> one contiguous 32B load per lane.
// ---------------------------------------------------------------------------
__device__ __forceinline__ v8f wmma_sw(const _Float16* A, int lda,
                                       const _Float16* Bp, int KB, int ct,
                                       v8f acc)
{
  const int lane = threadIdx.x & 31;
  const int m    = lane & 15;
  const int half = lane >> 4;
  for (int kb = 0; kb < KB; ++kb){
    const _Float16* pa = A + m*lda + kb*32 + half*8;
    v8h lo = *(const v8h*)pa;
    v8h hi = *(const v8h*)(pa + 16);
    v16h av = __builtin_shufflevector(lo, hi, 0,1,2,3,4,5,6,7,8,9,10,11,12,13,14,15);
    const _Float16* pb = Bp + ((size_t)(((ct*KB + kb)*2 + half)*16 + m) << 4);
    v16h bv = *(const v16h*)pb;
    acc = __builtin_amdgcn_wmma_f32_16x16x32_f16(false, av, false, bv,
                                                 (short)0, acc, false, false);
  }
  return acc;
}

// ---------------------------------------------------------------------------
__global__ void k_zero(float* p, long n){
  long i = (long)blockIdx.x*blockDim.x + threadIdx.x;
  long st = (long)gridDim.x*blockDim.x;
  for (; i < n; i += st) p[i] = 0.f;
}

// ---------------------------------------------------------------------------
// Weight prep: f32 -> f16 swizzled into WMMA B-fragment order.
// Fragment element tt (0..15) of (ct,kb,half,n):
//   k = kb*32 + (tt<8 ? half*8+tt : 16+half*8+(tt-8)),  col = ct*16+n
// ---------------------------------------------------------------------------
struct LayerW {
  const float *r1, *r2, *r3, *r4, *up;
  _Float16 *r1p, *r1T, *r2h, *r2T, *r3h, *r3T, *r4h, *r4T, *uph;
};

__device__ void swz(const float* src, int ld, int Kdim, int Ncols,
                    int kval, int nval, bool tr, _Float16* dst){
  int t = threadIdx.x, st = blockDim.x;
  int KB = Kdim >> 5, CT = Ncols >> 4;
  int total = CT*KB*512;
  for (int i = t; i < total; i += st){
    int tt   = i & 15;
    int n    = (i >> 4) & 15;
    int half = (i >> 8) & 1;
    int r    = i >> 9;
    int kb   = r % KB, ct = r / KB;
    int k   = kb*32 + ((tt < 8) ? (half*8 + tt) : (16 + half*8 + (tt - 8)));
    int col = ct*16 + n;
    float v = 0.f;
    if (k < kval && col < nval) v = tr ? src[col*ld + k] : src[k*ld + col];
    dst[i] = (_Float16)v;
  }
}

__device__ void prep_one(const LayerW w){
  swz(w.r1, 64, 32, 64, 8, 64, false, w.r1p);
  swz(w.r1, 64, 64, 16, 64, 8, true,  w.r1T);
  swz(w.r2, 64, 64, 64, 64, 64, false, w.r2h);
  swz(w.r2, 64, 64, 64, 64, 64, true,  w.r2T);
  swz(w.r3, 64, 64, 64, 64, 64, false, w.r3h);
  swz(w.r3, 64, 64, 64, 64, 64, true,  w.r3T);
  swz(w.up, 64, 64, 64, 64, 64, false, w.uph);
  swz(w.r4, 256, 64, 256, 64, 256, false, w.r4h);
  swz(w.r4, 256, 256, 64, 256, 64, true,  w.r4T);
}
__global__ void k_prep(LayerW a, LayerW b){ prep_one(a); prep_one(b); }

// ---------------------------------------------------------------------------
__global__ void k_embed(const float* attrs, const float* Wemb, const float* ae,
                        const int* batch, float* femb, float* out_e, int N){
  int i = blockIdx.x*blockDim.x + threadIdx.x;
  if (i >= N*64) return;
  int n = i >> 6, c = i & 63;
  int sp = speciesOf(attrs, n);
  femb[i] = Wemb[sp*64 + c];
  if (c < 3) atomicAdd(&out_e[batch[n]*3 + c], ae[sp]);   // e0 broadcast over 3 cols
}

__global__ void k_cvt16(const float* s, _Float16* d, long n){
  long i = (long)blockIdx.x*blockDim.x + threadIdx.x;
  long st = (long)gridDim.x*blockDim.x;
  for (; i < n; i += st) d[i] = (_Float16)s[i];
}

// fs = feats @ up  (node up-projection), wave per 16-node tile, WMMA
__global__ void __launch_bounds__(32)
k_up(const _Float16* feats16, const _Float16* uph, _Float16* fs16, int N){
  __shared__ _Float16 sT[16][64];
  __shared__ _Float16 sOut[16][64];
  int base = blockIdx.x*16;
  int tid = threadIdx.x, lane = tid & 31;
  for (int i = tid; i < 128; i += 32){
    int m = i >> 3, c8 = (i & 7)*8; int node = base + m;
    v8h v = {};
    if (node < N) v = *(const v8h*)(feats16 + (size_t)node*64 + c8);
    *(v8h*)(&sT[m][c8]) = v;
  }
  __syncthreads();
  int nn = lane & 15, half = lane >> 4;
  for (int ct = 0; ct < 4; ++ct){
    v8f acc = {};
    acc = wmma_sw(&sT[0][0], 64, uph, 2, ct, acc);
#pragma unroll
    for (int j = 0; j < 8; ++j){ int m = j + 8*half; sOut[m][ct*16+nn] = (_Float16)acc[j]; }
  }
  __syncthreads();
  for (int i = tid; i < 128; i += 32){
    int m = i >> 3, c8 = (i & 7)*8; int node = base + m;
    if (node < N) *(v8h*)(fs16 + (size_t)node*64 + c8) = *(const v8h*)(&sOut[m][c8]);
  }
}

// ---------------------------------------------------------------------------
// Edge geometry: unit vec, 16 sph harmonics, bessel*fcut (f16, zero-pad K=32),
// plus d(ef)/dr for the backward pass.
// ---------------------------------------------------------------------------
__global__ void k_geom(const float* pos, const float* shifts, const int* ei,
                       float* Y, float* U, float* Rinv, float* dEF,
                       _Float16* ef16, int E){
  int e = blockIdx.x*blockDim.x + threadIdx.x;
  if (e >= E) return;
  int sn = ei[e], rv = ei[E + e];
  float vx = pos[rv*3+0] - pos[sn*3+0] + shifts[e*3+0];
  float vy = pos[rv*3+1] - pos[sn*3+1] + shifts[e*3+1];
  float vz = pos[rv*3+2] - pos[sn*3+2] + shifts[e*3+2];
  float r2 = vx*vx + vy*vy + vz*vz + 1e-12f;
  float r  = sqrtf(r2);
  float ri = 1.f/r;
  float x = vx*ri, y = vy*ri, z = vz*ri;
  U[e*3+0]=x; U[e*3+1]=y; U[e*3+2]=z; Rinv[e]=ri;
  const float s3=1.7320508f, s5=2.236068f, s15=3.8729833f;
  const float s7=2.6457513f, s42=6.4807407f, s70=8.3666003f, s105=10.2469508f;
  float z2 = z*z;
  float Ys[16];
  Ys[0]=1.f; Ys[1]=s3*x; Ys[2]=s3*y; Ys[3]=s3*z;
  Ys[4]=s15*x*y; Ys[5]=s15*y*z; Ys[6]=0.5f*s5*(3.f*z2-1.f); Ys[7]=s15*x*z;
  Ys[8]=0.5f*s15*(x*x-y*y);
  Ys[9]=0.25f*s70*y*(3.f*x*x-y*y); Ys[10]=s105*x*y*z;
  Ys[11]=0.25f*s42*y*(5.f*z2-1.f); Ys[12]=0.5f*s7*z*(5.f*z2-3.f);
  Ys[13]=0.25f*s42*x*(5.f*z2-1.f); Ys[14]=0.5f*s105*z*(x*x-y*y);
  Ys[15]=0.25f*s70*x*(x*x-3.f*y*y);
#pragma unroll
  for (int i = 0; i < 16; ++i) Y[e*16+i] = Ys[i];
  float xx = r * (1.f/RMAXF);
  float fcut = 0.f, dfc = 0.f;
  if (xx < 1.f){
    float x4 = xx*xx*xx*xx;
    fcut = 1.f - 21.f*x4*xx + 35.f*x4*xx*xx - 15.f*x4*xx*xx*xx;
    dfc  = (-105.f*x4 + 210.f*x4*xx - 105.f*x4*xx*xx) * (1.f/RMAXF);
  }
  float amp = sqrtf(2.f/RMAXF);
#pragma unroll
  for (int b = 0; b < 8; ++b){
    float nb = (float)(b+1);
    float arg = nb*PI_F*xx;
    float sn_ = __sinf(arg), cs_ = __cosf(arg);
    float bess  = amp*sn_*ri;
    float dbess = amp*(nb*PI_F*(1.f/RMAXF)*cs_*ri - sn_*ri*ri);
    ef16[e*32+b] = (_Float16)(bess*fcut);
    dEF[e*8+b]   = dbess*fcut + bess*dfc;
  }
#pragma unroll
  for (int b = 8; b < 32; ++b) ef16[e*32+b] = (_Float16)0.f;
}

// ---------------------------------------------------------------------------
// Fused edge forward: radial MLP (WMMA chain) + message scatter into A.
// One wave per 16-edge tile.
// ---------------------------------------------------------------------------
__global__ void __launch_bounds__(32)
k_edge_fwd(const _Float16* ef16, const float* Y, const int* ei,
           const _Float16* fs16,
           const _Float16* r1p, const _Float16* r2h,
           const _Float16* r3h, const _Float16* r4h,
           float* Apre, int E){
  __shared__ _Float16 sEF[16][32];
  __shared__ _Float16 sH1[16][64];
  __shared__ _Float16 sH2[16][64];
  __shared__ int   sSend[16], sRecv[16];
  __shared__ float sY[16][16];
  int e0 = blockIdx.x * 16;
  int tid = threadIdx.x, lane = tid & 31;
  for (int i = tid; i < 16; i += 32){
    int e = e0 + i;
    sSend[i] = (e<E)? ei[e]     : -1;
    sRecv[i] = (e<E)? ei[E + e] : -1;
  }
  for (int i = tid; i < 64; i += 32){
    int m = i >> 2, c4 = (i & 3)*4; int e = e0 + m;
    float4 v = make_float4(0.f,0.f,0.f,0.f);
    if (e < E) v = *(const float4*)(Y + (size_t)e*16 + c4);
    *(float4*)(&sY[m][c4]) = v;
  }
  for (int i = tid; i < 64; i += 32){
    int m = i >> 2, c8 = (i & 3)*8; int e = e0 + m;
    v8h v = {};
    if (e < E) v = *(const v8h*)(ef16 + (size_t)e*32 + c8);
    *(v8h*)(&sEF[m][c8]) = v;
  }
  if (e0 + 16 < E) __builtin_prefetch(&ef16[(size_t)(e0+16)*32], 0, 0);
  __syncthreads();
  int nn = lane & 15, half = lane >> 4;
  // h1 = silu(ef @ r1)
  for (int ct = 0; ct < 4; ++ct){
    v8f acc = {};
    acc = wmma_sw(&sEF[0][0], 32, r1p, 1, ct, acc);
#pragma unroll
    for (int j = 0; j < 8; ++j){ int m=j+8*half; sH1[m][ct*16+nn] = (_Float16)siluf(acc[j]); }
  }
  __syncthreads();
  // h2 = silu(h1 @ r2)
  for (int ct = 0; ct < 4; ++ct){
    v8f acc = {};
    acc = wmma_sw(&sH1[0][0], 64, r2h, 2, ct, acc);
#pragma unroll
    for (int j = 0; j < 8; ++j){ int m=j+8*half; sH2[m][ct*16+nn] = (_Float16)siluf(acc[j]); }
  }
  __syncthreads();
  // h3 = silu(h2 @ r3)  -> back into sH1
  for (int ct = 0; ct < 4; ++ct){
    v8f acc = {};
    acc = wmma_sw(&sH2[0][0], 64, r3h, 2, ct, acc);
#pragma unroll
    for (int j = 0; j < 8; ++j){ int m=j+8*half; sH1[m][ct*16+nn] = (_Float16)siluf(acc[j]); }
  }
  __syncthreads();
  // tpw = h3 @ r4 (16 col tiles) + scatter  A[recv,s,k] += Y[e,s]*tpw*fs
  const int sBeg[4] = {0,1,4,9};
  const int sCnt[4] = {1,3,5,7};
  for (int ct = 0; ct < 16; ++ct){
    v8f acc = {};
    acc = wmma_sw(&sH1[0][0], 64, r4h, 2, ct, acc);
    int l  = ct >> 2;
    int kk = (ct & 3)*16 + nn;
#pragma unroll
    for (int j = 0; j < 8; ++j){
      int m   = j + 8*half;
      int snd = sSend[m];
      if (snd < 0) continue;
      int rcv = sRecv[m];
      float fsv = (float)fs16[(size_t)snd*64 + kk];
      float tv  = acc[j] * fsv;
      float* Ap = Apre + (size_t)rcv*1024 + kk;
      for (int q = 0; q < sCnt[l]; ++q){
        int ss = sBeg[l] + q;
        atomicAdd(Ap + ss*64, sY[m][ss] * tv);
      }
    }
  }
}

// ---------------------------------------------------------------------------
// A_post[n,s,d] = (A_pre[n,s,:] / 16) @ lin[L_MAP[s]]
__global__ void k_lin(const float* Apre, const float* lin, float* Apost, long NSD){
  long i = (long)blockIdx.x*blockDim.x + threadIdx.x;
  if (i >= NSD) return;
  int d = (int)(i & 63); int s = (int)((i >> 6) & 15); long n = i >> 10;
  const float* L = lin + c_lmap[s]*4096;
  const float* A = Apre + n*1024 + (long)s*64;
  float acc = 0.f;
  for (int c = 0; c < 64; ++c) acc += A[c] * L[c*64 + d];
  Apost[i] = acc * (1.f/16.f);
}

// Bs[n,k] from product-basis invariants
__global__ void k_bs(const float* Apost, const float* attrs, const float* wprod_s,
                     float* Bs, int N){
  int i = blockIdx.x*blockDim.x + threadIdx.x;
  if (i >= N*64) return;
  int n = i >> 6, k = i & 63;
  int sp = speciesOf(attrs, n);
  const float* W = wprod_s + (long)sp*8*64 + k;
  const float* A = Apost + (long)n*1024 + k;
  float a0 = A[0];
  float s1=0.f, s2=0.f, s3v=0.f;
  for (int s = 1; s <= 3;  ++s){ float v=A[s*64]; s1  += v*v; }
  for (int s = 4; s <= 8;  ++s){ float v=A[s*64]; s2  += v*v; }
  for (int s = 9; s <= 15; ++s){ float v=A[s*64]; s3v += v*v; }
  float w0=W[0],w1=W[64],w2=W[128],w3=W[192],w4=W[256],w5=W[320],w6=W[384],w7=W[448];
  Bs[i] = w0*a0 + w1*a0*a0 + w2*s1 + w3*s2 + w4*s3v
        + w5*a0*a0*a0 + w6*a0*s1 + w7*a0*s2;
}

// fout = Bs @ wpout + sc(fin, wsc[species])
__global__ void k_feats(const float* Bs, const float* wpout, const float* fin,
                        const float* wsc, const float* attrs, float* fout, int N){
  int i = blockIdx.x*blockDim.x + threadIdx.x;
  if (i >= N*64) return;
  int n = i >> 6, d = i & 63;
  int sp = speciesOf(attrs, n);
  float acc = 0.f;
  for (int k = 0; k < 64; ++k) acc += Bs[n*64+k] * wpout[k*64+d];
  const float* ws = wsc + (long)sp*4096 + d;
  for (int c = 0; c < 64; ++c) acc += fin[n*64+c] * ws[c*64];
  fout[i] = acc;
}

// feats_v[n,x,d] from vector product basis (layer0 only)
__global__ void k_vfeat(const float* Apost, const float* attrs, const float* wprod_v,
                        const float* wpout_v, float* featv, int N){
  long i = (long)blockIdx.x*blockDim.x + threadIdx.x;
  if (i >= (long)N*192) return;
  int n = (int)(i/192); int rem = (int)(i%192); int x = rem >> 6; int d = rem & 63;
  int sp = speciesOf(attrs, n);
  const float* Wv = wprod_v + (long)sp*256;
  const float* A  = Apost + (long)n*1024;
  float acc = 0.f;
  for (int k = 0; k < 64; ++k){
    float a0  = A[k];
    float a1x = A[(1+x)*64 + k];
    float v1=A[64+k], v2=A[128+k], v3=A[192+k];
    float s1 = v1*v1 + v2*v2 + v3*v3;
    float wv = Wv[k] + Wv[64+k]*a0 + Wv[128+k]*a0*a0 + Wv[192+k]*s1;
    acc += a1x * wv * wpout_v[k*64 + d];
  }
  featv[i] = acc;
}

// nacs (store) + dipoles (graph atomic)
__global__ void k_vread(const float* featv, const float* wr_n, const float* wr_d,
                        const int* batch, float* out_nacs, float* out_dip, int N){
  int i = blockIdx.x*blockDim.x + threadIdx.x;
  if (i >= N*9) return;
  int n = i/9; int tx = i%9; int t = tx/3; int x = tx%3;
  float an = 0.f, ad = 0.f;
  for (int c = 0; c < 64; ++c){
    float f = featv[(long)n*192 + x*64 + c];
    an += f * wr_n[c*3+t];
    ad += f * wr_d[c*3+t];
  }
  out_nacs[n*9 + t*3 + x] = an;
  atomicAdd(&out_dip[batch[n]*9 + t*3 + x], ad);
}

// layer0 readouts: En (3) + soc (45)
__global__ void k_read0(const float* f0, const float* wr_e, const float* wsoc,
                        const int* batch, float* out_e, float* out_s, int N){
  int i = blockIdx.x*blockDim.x + threadIdx.x;
  if (i >= N*48) return;
  int n = i/48; int j = i%48; int g = batch[n];
  float acc = 0.f;
  if (j < 3){
    for (int c = 0; c < 64; ++c) acc += f0[n*64+c]*wr_e[c*3+j];
    atomicAdd(&out_e[g*3+j], acc);
  } else {
    int t = j - 3;
    for (int c = 0; c < 64; ++c) acc += f0[n*64+c]*wsoc[c*45+t];
    atomicAdd(&out_s[g*45+t], acc);
  }
}

// layer1 readouts + grad of energy wrt f1
__global__ void k_read1(const float* f1, const float* wm1, const float* wm2,
                        const float* wms, const int* batch,
                        float* out_e, float* out_s, float* gf1, int N){
  int n = blockIdx.x*blockDim.x + threadIdx.x;
  if (n >= N) return;
  int g = batch[n];
  float z[16], h[16];
  for (int iH = 0; iH < 16; ++iH){
    float a = 0.f;
    for (int c = 0; c < 64; ++c) a += f1[n*64+c]*wm1[c*16+iH];
    z[iH] = a; h[iH] = siluf(a);
  }
  for (int t = 0; t < 3; ++t){
    float a = 0.f;
    for (int iH = 0; iH < 16; ++iH) a += h[iH]*wm2[iH*3+t];
    atomicAdd(&out_e[g*3+t], a);
  }
  for (int t = 0; t < 45; ++t){
    float a = 0.f;
    for (int iH = 0; iH < 16; ++iH) a += h[iH]*wms[iH*45+t];
    atomicAdd(&out_s[g*45+t], a);
  }
  float gz[16];
  for (int iH = 0; iH < 16; ++iH)
    gz[iH] = (wm2[iH*3]+wm2[iH*3+1]+wm2[iH*3+2]) * dsiluf(z[iH]);
  for (int c = 0; c < 64; ++c){
    float a = 0.f;
    for (int iH = 0; iH < 16; ++iH) a += gz[iH]*wm1[c*16+iH];
    gf1[n*64+c] = a;
  }
}

// grad wrt f0: direct wr_e term + wsc1 path
__global__ void k_gf0(const float* gf1, const float* wsc1, const float* attrs,
                      const float* wr_e, float* gf0, int N){
  int i = blockIdx.x*blockDim.x + threadIdx.x;
  if (i >= N*64) return;
  int n = i >> 6, c = i & 63;
  int sp = speciesOf(attrs, n);
  float acc = wr_e[c*3] + wr_e[c*3+1] + wr_e[c*3+2];
  const float* ws = wsc1 + (long)sp*4096 + c*64;
  for (int d = 0; d < 64; ++d) acc += gf1[n*64+d]*ws[d];
  gf0[i] = acc;
}

// gBs = gf @ wpout^T
__global__ void k_gbs(const float* gf, const float* wpout, float* gBs, int N){
  int i = blockIdx.x*blockDim.x + threadIdx.x;
  if (i >= N*64) return;
  int n = i >> 6, k = i & 63;
  float acc = 0.f;
  for (int d = 0; d < 64; ++d) acc += gf[n*64+d]*wpout[k*64+d];
  gBs[i] = acc;
}

// product-basis backward: gBs -> gA_post
__global__ void k_pbwd(const float* Apost, const float* attrs, const float* wprod_s,
                       const float* gBs, float* gTmp, int N){
  int i = blockIdx.x*blockDim.x + threadIdx.x;
  if (i >= N*64) return;
  int n = i >> 6, k = i & 63;
  int sp = speciesOf(attrs, n);
  const float* W = wprod_s + (long)sp*8*64 + k;
  const float* A = Apost + (long)n*1024 + k;
  float a0 = A[0];
  float s1=0.f, s2=0.f;
  for (int s = 1; s <= 3; ++s){ float v=A[s*64]; s1 += v*v; }
  for (int s = 4; s <= 8; ++s){ float v=A[s*64]; s2 += v*v; }
  float w0=W[0],w1=W[64],w2=W[128],w3=W[192],w4=W[256],w5=W[320],w6=W[384],w7=W[448];
  float gB = gBs[i];
  float* G = gTmp + (long)n*1024 + k;
  G[0] = gB*(w0 + 2.f*w1*a0 + 3.f*w5*a0*a0 + w6*s1 + w7*s2);
  float c1 = 2.f*gB*(w2 + w6*a0);
  float c2 = 2.f*gB*(w3 + w7*a0);
  float c3 = 2.f*gB*w4;
  for (int s = 1; s <= 3;  ++s) G[s*64] = c1*A[s*64];
  for (int s = 4; s <= 8;  ++s) G[s*64] = c2*A[s*64];
  for (int s = 9; s <= 15; ++s) G[s*64] = c3*A[s*64];
}

// gA_pre[n,s,c] = (gA_post[n,s,:] @ lin[l]^T) / 16
__global__ void k_linb(const float* gTmp, const float* lin, float* gA, long NSD){
  long i = (long)blockIdx.x*blockDim.x + threadIdx.x;
  if (i >= NSD) return;
  int c = (int)(i & 63); int s = (int)((i >> 6) & 15); long n = i >> 10;
  const float* L = lin + c_lmap[s]*4096 + c*64;
  const float* G = gTmp + n*1024 + (long)s*64;
  float acc = 0.f;
  for (int d = 0; d < 64; ++d) acc += G[d]*L[d];
  gA[i] = acc * (1.f/16.f);
}

// analytic gradient of the 16 sph components wrt unit vector
__device__ __forceinline__ void sph_grad(float x, float y, float z, const float* g,
                                         float& gx, float& gy, float& gz){
  const float s3=1.7320508f, s5=2.236068f, s15=3.8729833f;
  const float s7=2.6457513f, s42=6.4807407f, s70=8.3666003f, s105=10.2469508f;
  float z2 = z*z;
  gx = s3*g[1] + s15*y*g[4] + s15*z*g[7] + s15*x*g[8]
     + 1.5f*s70*x*y*g[9] + s105*y*z*g[10] + 0.25f*s42*(5.f*z2-1.f)*g[13]
     + s105*x*z*g[14] + 0.75f*s70*(x*x - y*y)*g[15];
  gy = s3*g[2] + s15*x*g[4] + s15*z*g[5] - s15*y*g[8]
     + 0.75f*s70*(x*x - y*y)*g[9] + s105*x*z*g[10] + 0.25f*s42*(5.f*z2-1.f)*g[11]
     - s105*y*z*g[14] - 1.5f*s70*x*y*g[15];
  gz = s3*g[3] + s15*y*g[5] + 3.f*s5*z*g[6] + s15*x*g[7]
     + s105*x*y*g[10] + 2.5f*s42*y*z*g[11] + 0.5f*s7*(15.f*z2-3.f)*g[12]
     + 2.5f*s42*x*z*g[13] + 0.5f*s105*(x*x-y*y)*g[14];
}

// ---------------------------------------------------------------------------
// Edge backward: recompute MLP in LDS, form g_tpw / g_Y from gA, run the
// transposed-weight WMMA chain down to g_ef, apply geometric chain rule and
// accumulate forces with global f32 atomics.
// ---------------------------------------------------------------------------
__global__ void __launch_bounds__(32)
k_edge_bwd(const _Float16* ef16, const float* Y, const float* U, const float* Rinv,
           const float* dEF, const int* ei, const _Float16* fs16,
           const _Float16* r1p, const _Float16* r2h,
           const _Float16* r3h, const _Float16* r4h,
           const _Float16* r1T, const _Float16* r2T,
           const _Float16* r3T, const _Float16* r4T,
           const float* gA, float* forces, int E)
{
  __shared__ _Float16 sEF[16][32];
  __shared__ _Float16 sP[16][64];
  __shared__ _Float16 sQ[16][64];
  __shared__ _Float16 sGT[16][256];
  __shared__ float sZ1[16][64], sZ2[16][64], sZ3[16][64];
  __shared__ float sGY[16][16];
  __shared__ float sGE[16][8];
  __shared__ int   sSend[16], sRecv[16];
  __shared__ float sY[16][16];

  int e0 = blockIdx.x * 16;
  int tid = threadIdx.x, lane = tid & 31;
  for (int i = tid; i < 16; i += 32){
    int e = e0 + i;
    sSend[i] = (e<E)? ei[e]     : -1;
    sRecv[i] = (e<E)? ei[E + e] : -1;
  }
  for (int i = tid; i < 64; i += 32){
    int m = i >> 2, c4 = (i & 3)*4; int e = e0 + m;
    float4 v = make_float4(0.f,0.f,0.f,0.f);
    if (e < E) v = *(const float4*)(Y + (size_t)e*16 + c4);
    *(float4*)(&sY[m][c4]) = v;
  }
  for (int i = tid; i < 256; i += 32) ((float*)sGY)[i] = 0.f;
  for (int i = tid; i < 64; i += 32){
    int m = i >> 2, c8 = (i & 3)*8; int e = e0 + m;
    v8h v = {};
    if (e < E) v = *(const v8h*)(ef16 + (size_t)e*32 + c8);
    *(v8h*)(&sEF[m][c8]) = v;
  }
  if (e0 + 16 < E) __builtin_prefetch(&ef16[(size_t)(e0+16)*32], 0, 0);
  __syncthreads();
  int nn = lane & 15, half = lane >> 4;

  // forward recompute: h1 -> sP (z1), h2 -> sQ (z2), h3 -> sP (z3)
  for (int ct = 0; ct < 4; ++ct){
    v8f acc = {};
    acc = wmma_sw(&sEF[0][0], 32, r1p, 1, ct, acc);
#pragma unroll
    for (int j = 0; j < 8; ++j){ int m=j+8*half; int col=ct*16+nn;
      sZ1[m][col]=acc[j]; sP[m][col]=(_Float16)siluf(acc[j]); }
  }
  __syncthreads();
  for (int ct = 0; ct < 4; ++ct){
    v8f acc = {};
    acc = wmma_sw(&sP[0][0], 64, r2h, 2, ct, acc);
#pragma unroll
    for (int j = 0; j < 8; ++j){ int m=j+8*half; int col=ct*16+nn;
      sZ2[m][col]=acc[j]; sQ[m][col]=(_Float16)siluf(acc[j]); }
  }
  __syncthreads();
  for (int ct = 0; ct < 4; ++ct){
    v8f acc = {};
    acc = wmma_sw(&sQ[0][0], 64, r3h, 2, ct, acc);
#pragma unroll
    for (int j = 0; j < 8; ++j){ int m=j+8*half; int col=ct*16+nn;
      sZ3[m][col]=acc[j]; sP[m][col]=(_Float16)siluf(acc[j]); }
  }
  __syncthreads();

  // tpw tiles (h3 @ r4) -> g_tpw (sGT) and gY (LDS atomics)
  const int sBeg[4] = {0,1,4,9};
  const int sCnt[4] = {1,3,5,7};
  for (int ct = 0; ct < 16; ++ct){
    v8f acc = {};
    acc = wmma_sw(&sP[0][0], 64, r4h, 2, ct, acc);
    int l  = ct >> 2;
    int kk = (ct & 3)*16 + nn;
#pragma unroll
    for (int j = 0; j < 8; ++j){
      int m   = j + 8*half;
      int snd = sSend[m];
      float gt = 0.f;
      if (snd >= 0){
        int rcv = sRecv[m];
        float fsv = (float)fs16[(size_t)snd*64 + kk];
        const float* gap = gA + (size_t)rcv*1024 + kk;
        float tpwv = acc[j];
        for (int q = 0; q < sCnt[l]; ++q){
          int ss = sBeg[l] + q;
          float g = gap[ss*64];
          gt += sY[m][ss] * g;
          atomicAdd(&sGY[m][ss], g * tpwv * fsv);
        }
        gt *= fsv;
      }
      sGT[m][ct*16+nn] = (_Float16)gt;
    }
  }
  __syncthreads();

  // g_h3 = g_tpw @ r4^T, * silu'(z3) -> sQ
  for (int ct = 0; ct < 4; ++ct){
    v8f acc = {};
    acc = wmma_sw(&sGT[0][0], 256, r4T, 8, ct, acc);
#pragma unroll
    for (int j = 0; j < 8; ++j){ int m=j+8*half; int col=ct*16+nn;
      sQ[m][col] = (_Float16)(acc[j]*dsiluf(sZ3[m][col])); }
  }
  __syncthreads();
  // g_h2 = gz3 @ r3^T, * silu'(z2) -> sP
  for (int ct = 0; ct < 4; ++ct){
    v8f acc = {};
    acc = wmma_sw(&sQ[0][0], 64, r3T, 2, ct, acc);
#pragma unroll
    for (int j = 0; j < 8; ++j){ int m=j+8*half; int col=ct*16+nn;
      sP[m][col] = (_Float16)(acc[j]*dsiluf(sZ2[m][col])); }
  }
  __syncthreads();
  // g_h1 = gz2 @ r2^T, * silu'(z1) -> sQ
  for (int ct = 0; ct < 4; ++ct){
    v8f acc = {};
    acc = wmma_sw(&sP[0][0], 64, r2T, 2, ct, acc);
#pragma unroll
    for (int j = 0; j < 8; ++j){ int m=j+8*half; int col=ct*16+nn;
      sQ[m][col] = (_Float16)(acc[j]*dsiluf(sZ1[m][col])); }
  }
  __syncthreads();
  // g_ef = gz1 @ r1^T  (only cols 0..7 meaningful)
  {
    v8f acc = {};
    acc = wmma_sw(&sQ[0][0], 64, r1T, 2, 0, acc);
    if (nn < 8){
#pragma unroll
      for (int j = 0; j < 8; ++j){ int m=j+8*half; sGE[m][nn] = acc[j]; }
    }
  }
  __syncthreads();

  // per-edge finalize on lanes 0..15
  if (lane < 16){
    int e = e0 + lane;
    if (e < E && sSend[lane] >= 0){
      float gr = 0.f;
#pragma unroll
      for (int b = 0; b < 8; ++b) gr += sGE[lane][b]*dEF[(size_t)e*8+b];
      float gY16[16];
#pragma unroll
      for (int s = 0; s < 16; ++s) gY16[s] = sGY[lane][s];
      float ux=U[e*3+0], uy=U[e*3+1], uz=U[e*3+2], ri=Rinv[e];
      float gux, guy, guz;
      sph_grad(ux, uy, uz, gY16, gux, guy, guz);
      float dot = gux*ux + guy*uy + guz*uz;
      float gvx = gr*ux + ri*(gux - dot*ux);
      float gvy = gr*uy + ri*(guy - dot*uy);
      float gvz = gr*uz + ri*(guz - dot*uz);
      int snd = sSend[lane], rcv = sRecv[lane];
      atomicAdd(&forces[rcv*3+0], -gvx);
      atomicAdd(&forces[rcv*3+1], -gvy);
      atomicAdd(&forces[rcv*3+2], -gvz);
      atomicAdd(&forces[snd*3+0],  gvx);
      atomicAdd(&forces[snd*3+1],  gvy);
      atomicAdd(&forces[snd*3+2],  gvz);
    }
  }
}

// ---------------------------------------------------------------------------
extern "C" void kernel_launch(void* const* d_in, const int* in_sizes, int n_in,
                              void* d_out, int out_size, void* d_ws, size_t ws_size,
                              hipStream_t stream) {
  const float* pos    = (const float*)d_in[0];
  const float* attrs  = (const float*)d_in[1];
  const float* shifts = (const float*)d_in[2];
  const int*   ei     = (const int*)d_in[3];
  const int*   batch  = (const int*)d_in[4];
  const float* ae     = (const float*)d_in[6];
  const float* Wemb   = (const float*)d_in[7];
  // layer0
  const float *l0r1=(const float*)d_in[8],  *l0r2=(const float*)d_in[9];
  const float *l0r3=(const float*)d_in[10], *l0r4=(const float*)d_in[11];
  const float *l0up=(const float*)d_in[12], *l0lin=(const float*)d_in[13];
  const float *l0wsc=(const float*)d_in[14], *l0wps=(const float*)d_in[15];
  const float *l0wpv=(const float*)d_in[16], *l0pos_=(const float*)d_in[17];
  const float *l0pov=(const float*)d_in[18], *l0wre=(const float*)d_in[19];
  const float *l0wrn=(const float*)d_in[20], *l0wrd=(const float*)d_in[21];
  const float *l0soc=(const float*)d_in[22];
  // layer1
  const float *l1r1=(const float*)d_in[23], *l1r2=(const float*)d_in[24];
  const float *l1r3=(const float*)d_in[25], *l1r4=(const float*)d_in[26];
  const float *l1up=(const float*)d_in[27], *l1lin=(const float*)d_in[28];
  const float *l1wsc=(const float*)d_in[29], *l1wps=(const float*)d_in[30];
  const float *l1pos_=(const float*)d_in[31], *l1wm1=(const float*)d_in[32];
  const float *l1wm2=(const float*)d_in[33], *l1wms=(const float*)d_in[34];

  const int N = in_sizes[0]/3;
  const int E = in_sizes[2]/3;
  const int G = (out_size - 12*N)/57;   // out = 3G+3N+9G+9N+45G

  float* out   = (float*)d_out;
  float* out_e = out;                    // [G,3]
  float* out_f = out + 3*G;              // [N,3]
  float* out_d = out + 3*G + 3*N;        // [G,3,3]
  float* out_n = out + 12*G + 3*N;       // [N,3,3]
  float* out_s = out + 12*G + 12*N;      // [G,45]

  // ---- workspace carve ----
  char* wsb = (char*)d_ws; size_t off = 0;
  auto carve = [&](size_t bytes)->char*{
    char* p = wsb + off; off = (off + bytes + 255) & ~(size_t)255; return p;
  };
  auto carveH = [&](size_t n)->_Float16*{ return (_Float16*)carve(n*2); };
  auto carveF = [&](size_t n)->float*   { return (float*)carve(n*4); };

  LayerW W0, W1;
  W0.r1=l0r1; W0.r2=l0r2; W0.r3=l0r3; W0.r4=l0r4; W0.up=l0up;
  W1.r1=l1r1; W1.r2=l1r2; W1.r3=l1r3; W1.r4=l1r4; W1.up=l1up;
  W0.r1p=carveH(2048);  W0.r1T=carveH(1024);
  W0.r2h=carveH(4096);  W0.r2T=carveH(4096);
  W0.r3h=carveH(4096);  W0.r3T=carveH(4096);
  W0.r4h=carveH(16384); W0.r4T=carveH(16384);
  W0.uph=carveH(4096);
  W1.r1p=carveH(2048);  W1.r1T=carveH(1024);
  W1.r2h=carveH(4096);  W1.r2T=carveH(4096);
  W1.r3h=carveH(4096);  W1.r3T=carveH(4096);
  W1.r4h=carveH(16384); W1.r4T=carveH(16384);
  W1.uph=carveH(4096);

  _Float16* femb16 = carveH((size_t)N*64);
  _Float16* f016   = carveH((size_t)N*64);
  _Float16* fs0    = carveH((size_t)N*64);
  _Float16* fs1    = carveH((size_t)N*64);
  _Float16* ef16   = carveH((size_t)E*32);

  float* femb = carveF((size_t)N*64);
  float* f0   = carveF((size_t)N*64);
  float* f1   = carveF((size_t)N*64);
  float* Bs   = carveF((size_t)N*64);
  float* gf1  = carveF((size_t)N*64);
  float* gf0  = carveF((size_t)N*64);
  float* gBs  = carveF((size_t)N*64);
  float* Yb   = carveF((size_t)E*16);
  float* Ub   = carveF((size_t)E*3);
  float* Rib  = carveF((size_t)E);
  float* dEFb = carveF((size_t)E*8);
  float* Apre   = carveF((size_t)N*1024);
  float* Apost0 = carveF((size_t)N*1024);
  float* Apost1 = carveF((size_t)N*1024);
  float* gTmp   = carveF((size_t)N*1024);
  float* gAb    = carveF((size_t)N*1024);
  float* featv  = carveF((size_t)N*192);
  (void)ws_size; (void)n_in; (void)l0pov;

  const int B = 256;
  auto gr = [&](long n){ return (int)((n + B - 1)/B); };
  const long NSD   = (long)N*1024;
  const int  tiles = (E + 15)/16;
  const int  ntile = (N + 15)/16;

  // ---- forward ----
  k_zero<<<gr(out_size), B, 0, stream>>>(out, out_size);
  k_prep<<<1, 256, 0, stream>>>(W0, W1);
  k_embed<<<gr((long)N*64), B, 0, stream>>>(attrs, Wemb, ae, batch, femb, out_e, N);
  k_cvt16<<<gr((long)N*64), B, 0, stream>>>(femb, femb16, (long)N*64);
  k_up<<<ntile, 32, 0, stream>>>(femb16, W0.uph, fs0, N);
  k_geom<<<gr(E), B, 0, stream>>>(pos, shifts, ei, Yb, Ub, Rib, dEFb, ef16, E);

  k_zero<<<gr(NSD), B, 0, stream>>>(Apre, NSD);
  k_edge_fwd<<<tiles, 32, 0, stream>>>(ef16, Yb, ei, fs0,
                                       W0.r1p, W0.r2h, W0.r3h, W0.r4h, Apre, E);
  k_lin<<<gr(NSD), B, 0, stream>>>(Apre, l0lin, Apost0, NSD);
  k_bs<<<gr((long)N*64), B, 0, stream>>>(Apost0, attrs, l0wps, Bs, N);
  k_feats<<<gr((long)N*64), B, 0, stream>>>(Bs, l0pos_, femb, l0wsc, attrs, f0, N);
  k_vfeat<<<gr((long)N*192), B, 0, stream>>>(Apost0, attrs, l0wpv, l0pov, featv, N);
  k_vread<<<gr((long)N*9), B, 0, stream>>>(featv, l0wrn, l0wrd, batch, out_n, out_d, N);
  k_read0<<<gr((long)N*48), B, 0, stream>>>(f0, l0wre, l0soc, batch, out_e, out_s, N);

  k_cvt16<<<gr((long)N*64), B, 0, stream>>>(f0, f016, (long)N*64);
  k_up<<<ntile, 32, 0, stream>>>(f016, W1.uph, fs1, N);
  k_zero<<<gr(NSD), B, 0, stream>>>(Apre, NSD);
  k_edge_fwd<<<tiles, 32, 0, stream>>>(ef16, Yb, ei, fs1,
                                       W1.r1p, W1.r2h, W1.r3h, W1.r4h, Apre, E);
  k_lin<<<gr(NSD), B, 0, stream>>>(Apre, l1lin, Apost1, NSD);
  k_bs<<<gr((long)N*64), B, 0, stream>>>(Apost1, attrs, l1wps, Bs, N);
  k_feats<<<gr((long)N*64), B, 0, stream>>>(Bs, l1pos_, f0, l1wsc, attrs, f1, N);
  k_read1<<<gr(N), B, 0, stream>>>(f1, l1wm1, l1wm2, l1wms, batch, out_e, out_s, gf1, N);

  // ---- backward (forces) ----
  // layer1 path
  k_gbs<<<gr((long)N*64), B, 0, stream>>>(gf1, l1pos_, gBs, N);
  k_pbwd<<<gr((long)N*64), B, 0, stream>>>(Apost1, attrs, l1wps, gBs, gTmp, N);
  k_linb<<<gr(NSD), B, 0, stream>>>(gTmp, l1lin, gAb, NSD);
  k_edge_bwd<<<tiles, 32, 0, stream>>>(ef16, Yb, Ub, Rib, dEFb, ei, fs1,
                                       W1.r1p, W1.r2h, W1.r3h, W1.r4h,
                                       W1.r1T, W1.r2T, W1.r3T, W1.r4T,
                                       gAb, out_f, E);
  // layer0 path
  k_gf0<<<gr((long)N*64), B, 0, stream>>>(gf1, l1wsc, attrs, l0wre, gf0, N);
  k_gbs<<<gr((long)N*64), B, 0, stream>>>(gf0, l0pos_, gBs, N);
  k_pbwd<<<gr((long)N*64), B, 0, stream>>>(Apost0, attrs, l0wps, gBs, gTmp, N);
  k_linb<<<gr(NSD), B, 0, stream>>>(gTmp, l0lin, gAb, NSD);
  k_edge_bwd<<<tiles, 32, 0, stream>>>(ef16, Yb, Ub, Rib, dEFb, ei, fs0,
                                       W0.r1p, W0.r2h, W0.r3h, W0.r4h,
                                       W0.r1T, W0.r2T, W0.r3T, W0.r4T,
                                       gAb, out_f, E);
}
